// HGTLayer_53188874994368
// MI455X (gfx1250) — matrix operand order repaired and checked by reading, not compiled, not run on hardware
//
#include <hip/hip_runtime.h>
#include <hip/hip_bf16.h>
#include <math.h>

#define N_NODE 50000
#define DIM 128
#define NH 4
#define DKH 32
#define NEDGE 600000
#define LDSW_STRIDE 136   // 128 + pad, in f16 elements

typedef __attribute__((ext_vector_type(16))) _Float16 v16h;
typedef __attribute__((ext_vector_type(8)))  float    v8f;

__device__ __forceinline__ unsigned flip_f32(float f) {
    unsigned u = __float_as_uint(f);
    return u ^ ((unsigned)((int)u >> 31) | 0x80000000u);
}
__device__ __forceinline__ float unflip_f32(unsigned k) {
    unsigned u = (k & 0x80000000u) ? (k ^ 0x80000000u) : ~k;
    return __uint_as_float(u);
}
#define FLIP_NEG_INF 0x007FFFFFu

// ---------------------------------------------------------------------------
// Fold per-head transform T[h][k][j] (H x 32 x 32) into projection weight:
//   Wc[i, h*32+j] = sum_k W[i, h*32+k] * T[h,k,j];  bc likewise from b.
// ---------------------------------------------------------------------------
__global__ void fold_kernel(const float* __restrict__ W, const float* __restrict__ b,
                            const float* __restrict__ T,
                            float* __restrict__ Wc, float* __restrict__ bc) {
    int gid = blockIdx.x * blockDim.x + threadIdx.x;
    if (gid < DIM * DIM) {
        int i = gid >> 7, c = gid & 127, h = c >> 5, j = c & 31;
        const float* wr = W + i * DIM + h * DKH;
        const float* tc = T + h * DKH * DKH + j;
        float s = 0.f;
#pragma unroll
        for (int k = 0; k < DKH; ++k) s += wr[k] * tc[k * DKH];
        Wc[gid] = s;
    }
    if (gid < DIM) {
        int h = gid >> 5, j = gid & 31;
        const float* tc = T + h * DKH * DKH + j;
        float s = 0.f;
#pragma unroll
        for (int k = 0; k < DKH; ++k) s += b[h * DKH + k] * tc[k * DKH];
        bc[gid] = s;
    }
}

// ---------------------------------------------------------------------------
// WMMA GEMM: out[N,128] = X[N,128] @ W[128,128] + bias (MODE 0)
// MODE 1: out = alpha*(X@W+bias) + (1-alpha)*resid, alpha = sigmoid(skip[0])
// One wave computes a 16x128 slab via 8x v_wmma_f32_16x16x32_f16.
// ---------------------------------------------------------------------------
template <int MODE>
__global__ void __launch_bounds__(256) gemm_wmma_kernel(
    const float* __restrict__ X, const float* __restrict__ W,
    const float* __restrict__ bias, const float* __restrict__ resid,
    const float* __restrict__ skip, float* __restrict__ out, int nrows) {
    __shared__ _Float16 ldsW[DIM * LDSW_STRIDE];  // W^T in f16, padded stride

    // Stage W transposed into LDS as f16 (coalesced global reads).
    for (int idx = threadIdx.x; idx < DIM * DIM; idx += blockDim.x) {
        int k = idx >> 7, n = idx & 127;
        ldsW[n * LDSW_STRIDE + k] = (_Float16)W[idx];
    }
    __syncthreads();

    const int wave = threadIdx.x >> 5;
    const int lane = threadIdx.x & 31;
    const int rowbase = (blockIdx.x * 8 + wave) * 16;
    if (rowbase >= nrows) return;

    const int lhalf = lane >> 4;   // 0: lanes 0-15, 1: lanes 16-31
    const int lmod  = lane & 15;
    const int arow  = rowbase + lmod;

    float alpha = 0.f;
    if constexpr (MODE == 1) alpha = 1.f / (1.f + __expf(-skip[0]));

    v8f acc[8] = {};

    for (int k0 = 0; k0 < DIM; k0 += 32) {
        // ---- A fragment: lane holds row (rowbase+lmod), K runs
        //      [k0+lhalf*8 .. +7] and [k0+16+lhalf*8 .. +7]
        const float* xr = X + (size_t)arow * DIM + k0 + lhalf * 8;
        float4 f0 = *(const float4*)(xr);
        float4 f1 = *(const float4*)(xr + 4);
        float4 f2 = *(const float4*)(xr + 16);
        float4 f3 = *(const float4*)(xr + 20);
        v16h a;
        a[0]=(_Float16)f0.x; a[1]=(_Float16)f0.y; a[2]=(_Float16)f0.z; a[3]=(_Float16)f0.w;
        a[4]=(_Float16)f1.x; a[5]=(_Float16)f1.y; a[6]=(_Float16)f1.z; a[7]=(_Float16)f1.w;
        a[8]=(_Float16)f2.x; a[9]=(_Float16)f2.y; a[10]=(_Float16)f2.z; a[11]=(_Float16)f2.w;
        a[12]=(_Float16)f3.x; a[13]=(_Float16)f3.y; a[14]=(_Float16)f3.z; a[15]=(_Float16)f3.w;

#pragma unroll
        for (int t = 0; t < 8; ++t) {
            // ---- B fragment: lane holds column N=t*16+lmod,
            //      K [k0+lhalf*8 .. +7] and [k0+16+lhalf*8 .. +7] (from W^T LDS)
            const _Float16* bp = ldsW + (t * 16 + lmod) * LDSW_STRIDE + k0 + lhalf * 8;
            union { v16h v; uint4 q[2]; } bu;
            bu.q[0] = *(const uint4*)(bp);
            bu.q[1] = *(const uint4*)(bp + 16);
            acc[t] = __builtin_amdgcn_wmma_f32_16x16x32_f16(
                false, a, false, bu.v, (short)0, acc[t], false, false);
        }
    }

    // ---- Epilogue. C layout: acc[t][r] -> row rowbase + r + lhalf*8,
    //      col t*16 + lmod.
#pragma unroll
    for (int t = 0; t < 8; ++t) {
        const int col = t * 16 + lmod;
        const float bv = bias[col];
#pragma unroll
        for (int r = 0; r < 8; ++r) {
            const int row = rowbase + lhalf * 8 + r;
            float val = acc[t][r] + bv;
            if constexpr (MODE == 1)
                val = alpha * val + (1.f - alpha) * resid[(size_t)row * DIM + col];
            out[(size_t)row * DIM + col] = val;
        }
    }
}

// ---------------------------------------------------------------------------
// Init per-destination softmax state + zero accumulator.
// ---------------------------------------------------------------------------
__global__ void init_kernel(unsigned* __restrict__ maxBuf, float* __restrict__ denom,
                            float* __restrict__ hbuf, int n) {
    int idx = blockIdx.x * blockDim.x + threadIdx.x;
    if (idx < n * DIM) hbuf[idx] = 0.f;
    if (idx < n * NH) { maxBuf[idx] = FLIP_NEG_INF; denom[idx] = 0.f; }
}

// ---------------------------------------------------------------------------
// Pass 1: per (edge, head) score = mu[h]/sqrt(dk) * <q[dst,h], k[src,h]>,
// 8 lanes per (edge,head); atomicMax on flipped-uint keys for segment max.
// ---------------------------------------------------------------------------
__global__ void __launch_bounds__(256) score_kernel(
    const float* __restrict__ Q, const float* __restrict__ K,
    const int* __restrict__ src, const int* __restrict__ dst,
    const float* __restrict__ mu,
    float* __restrict__ scoreBuf, unsigned* __restrict__ maxBuf) {
    long long gid = (long long)blockIdx.x * blockDim.x + threadIdx.x;
    long long group = gid >> 3;
    int l = (int)(gid & 7);
    if (group >= (long long)NEDGE * NH) return;
    int e = (int)(group >> 2);
    int h = (int)(group & 3);
    int s = src[e], d = dst[e];

    float4 qv = *(const float4*)(Q + (size_t)d * DIM + h * DKH + l * 4);
    float4 kv = *(const float4*)(K + (size_t)s * DIM + h * DKH + l * 4);
    float p = qv.x * kv.x + qv.y * kv.y + qv.z * kv.z + qv.w * kv.w;
    p += __shfl_xor(p, 4, 8);
    p += __shfl_xor(p, 2, 8);
    p += __shfl_xor(p, 1, 8);
    if (l == 0) {
        float sc = p * mu[h] * 0.17677669529663688f;  // 1/sqrt(32)
        scoreBuf[(size_t)e * NH + h] = sc;
        atomicMax(&maxBuf[(size_t)d * NH + h], flip_f32(sc));
    }
}

// ---------------------------------------------------------------------------
// Pass 2: w = exp(score - max[dst]); denom[dst] += w; h[dst] += w * v[src].
// ---------------------------------------------------------------------------
__global__ void __launch_bounds__(256) agg_kernel(
    const float* __restrict__ V,
    const int* __restrict__ src, const int* __restrict__ dst,
    const float* __restrict__ scoreBuf, const unsigned* __restrict__ maxBuf,
    float* __restrict__ denom, float* __restrict__ hbuf) {
    long long gid = (long long)blockIdx.x * blockDim.x + threadIdx.x;
    long long group = gid >> 3;
    int l = (int)(gid & 7);
    if (group >= (long long)NEDGE * NH) return;
    int e = (int)(group >> 2);
    int h = (int)(group & 3);
    int s = src[e], d = dst[e];

    float sc = scoreBuf[(size_t)e * NH + h];
    float m = unflip_f32(maxBuf[(size_t)d * NH + h]);
    float w = __expf(sc - m);
    if (l == 0) atomicAdd(&denom[(size_t)d * NH + h], w);

    float4 vv = *(const float4*)(V + (size_t)s * DIM + h * DKH + l * 4);
    float* hp = hbuf + (size_t)d * DIM + h * DKH + l * 4;
    atomicAdd(hp + 0, w * vv.x);
    atomicAdd(hp + 1, w * vv.y);
    atomicAdd(hp + 2, w * vv.z);
    atomicAdd(hp + 3, w * vv.w);
}

// ---------------------------------------------------------------------------
// h /= denom (per head), guarding empty segments.
// ---------------------------------------------------------------------------
__global__ void norm_h_kernel(float* __restrict__ hbuf, const float* __restrict__ denom, int n) {
    int idx = blockIdx.x * blockDim.x + threadIdx.x;
    if (idx >= n * DIM) return;
    int node = idx >> 7, h = (idx >> 5) & 3;
    float d = denom[node * NH + h];
    float r = (d != 0.f) ? 1.f / d : 0.f;
    hbuf[idx] *= r;
}

// ---------------------------------------------------------------------------
// In-place LayerNorm over last dim (128); one wave per row, 8 rows/block.
// ---------------------------------------------------------------------------
__global__ void __launch_bounds__(256) ln_kernel(
    float* __restrict__ out, const float* __restrict__ g, const float* __restrict__ b,
    int nrows) {
    int row = blockIdx.x * 8 + (threadIdx.x >> 5);
    int lane = threadIdx.x & 31;
    if (row >= nrows) return;
    float* rp = out + (size_t)row * DIM + lane * 4;
    float4 x = *(const float4*)rp;
    float s = x.x + x.y + x.z + x.w;
#pragma unroll
    for (int m = 16; m; m >>= 1) s += __shfl_xor(s, m, 32);
    float mean = s * (1.f / 128.f);
    float dx = x.x - mean, dy = x.y - mean, dz = x.z - mean, dw = x.w - mean;
    float ss = dx * dx + dy * dy + dz * dz + dw * dw;
#pragma unroll
    for (int m = 16; m; m >>= 1) ss += __shfl_xor(ss, m, 32);
    float inv = rsqrtf(ss * (1.f / 128.f) + 1e-5f);
    float4 gg = *(const float4*)(g + lane * 4);
    float4 bb = *(const float4*)(b + lane * 4);
    x.x = dx * inv * gg.x + bb.x;
    x.y = dy * inv * gg.y + bb.y;
    x.z = dz * inv * gg.z + bb.z;
    x.w = dw * inv * gg.w + bb.w;
    *(float4*)rp = x;
}

// ---------------------------------------------------------------------------
extern "C" void kernel_launch(void* const* d_in, const int* in_sizes, int n_in,
                              void* d_out, int out_size, void* d_ws, size_t ws_size,
                              hipStream_t stream) {
    const float* feats_user = (const float*)d_in[0];
    const float* feats_item = (const float*)d_in[1];
    const int* src_ui = (const int*)d_in[2];
    const int* dst_ui = (const int*)d_in[3];
    const int* src_iu = (const int*)d_in[4];
    const int* dst_iu = (const int*)d_in[5];
    const float *Wk_u = (const float*)d_in[6],  *bk_u = (const float*)d_in[7];
    const float *Wq_u = (const float*)d_in[8],  *bq_u = (const float*)d_in[9];
    const float *Wv_u = (const float*)d_in[10], *bv_u = (const float*)d_in[11];
    const float *Wa_u = (const float*)d_in[12], *ba_u = (const float*)d_in[13];
    const float *lng_u = (const float*)d_in[14], *lnb_u = (const float*)d_in[15];
    const float *skip_u = (const float*)d_in[16];
    const float *Wk_i = (const float*)d_in[17], *bk_i = (const float*)d_in[18];
    const float *Wq_i = (const float*)d_in[19], *bq_i = (const float*)d_in[20];
    const float *Wv_i = (const float*)d_in[21], *bv_i = (const float*)d_in[22];
    const float *Wa_i = (const float*)d_in[23], *ba_i = (const float*)d_in[24];
    const float *lng_i = (const float*)d_in[25], *lnb_i = (const float*)d_in[26];
    const float *skip_i = (const float*)d_in[27];
    const float *mu_ui = (const float*)d_in[28], *watt_ui = (const float*)d_in[29],
                *wmsg_ui = (const float*)d_in[30];
    const float *mu_iu = (const float*)d_in[31], *watt_iu = (const float*)d_in[32],
                *wmsg_iu = (const float*)d_in[33];

    float* out_user = (float*)d_out;
    float* out_item = out_user + (size_t)N_NODE * DIM;

    // Workspace carve (floats), reused across the two sequential etypes.
    float* ws = (float*)d_ws;
    size_t o = 0;
    float* Wc_k = ws + o; o += DIM * DIM;
    float* bc_k = ws + o; o += DIM;
    float* Wc_v = ws + o; o += DIM * DIM;
    float* bc_v = ws + o; o += DIM;
    o = (o + 63) & ~(size_t)63;
    float* Kbuf = ws + o; o += (size_t)N_NODE * DIM;
    float* Vbuf = ws + o; o += (size_t)N_NODE * DIM;
    float* Qbuf = ws + o; o += (size_t)N_NODE * DIM;
    float* scoreBuf = ws + o; o += (size_t)NEDGE * NH;
    unsigned* maxBuf = (unsigned*)(ws + o); o += (size_t)N_NODE * NH;
    float* denomBuf = ws + o; o += (size_t)N_NODE * NH;
    float* h_ui = ws + o; o += (size_t)N_NODE * DIM;   // aggregated onto items
    float* h_iu = ws + o; o += (size_t)N_NODE * DIM;   // aggregated onto users

    const int GEMM_BLOCKS = (N_NODE + 127) / 128;               // 391
    const int ELEM_BLOCKS = (N_NODE * DIM + 255) / 256;         // 25000
    const int EDGE_BLOCKS = (int)(((long long)NEDGE * NH * 8 + 255) / 256);  // 75000
    const int LN_BLOCKS = (N_NODE + 7) / 8;                     // 6250

    // ---------------- etype 'ui' : user -> item -----------------------------
    fold_kernel<<<64, 256, 0, stream>>>(Wk_u, bk_u, watt_ui, Wc_k, bc_k);
    fold_kernel<<<64, 256, 0, stream>>>(Wv_u, bv_u, wmsg_ui, Wc_v, bc_v);
    gemm_wmma_kernel<0><<<GEMM_BLOCKS, 256, 0, stream>>>(feats_user, Wc_k, bc_k,
        nullptr, nullptr, Kbuf, N_NODE);
    gemm_wmma_kernel<0><<<GEMM_BLOCKS, 256, 0, stream>>>(feats_user, Wc_v, bc_v,
        nullptr, nullptr, Vbuf, N_NODE);
    gemm_wmma_kernel<0><<<GEMM_BLOCKS, 256, 0, stream>>>(feats_item, Wq_i, bq_i,
        nullptr, nullptr, Qbuf, N_NODE);
    init_kernel<<<ELEM_BLOCKS, 256, 0, stream>>>(maxBuf, denomBuf, h_ui, N_NODE);
    score_kernel<<<EDGE_BLOCKS, 256, 0, stream>>>(Qbuf, Kbuf, src_ui, dst_ui, mu_ui,
        scoreBuf, maxBuf);
    agg_kernel<<<EDGE_BLOCKS, 256, 0, stream>>>(Vbuf, src_ui, dst_ui, scoreBuf, maxBuf,
        denomBuf, h_ui);
    norm_h_kernel<<<ELEM_BLOCKS, 256, 0, stream>>>(h_ui, denomBuf, N_NODE);

    // ---------------- etype 'iu' : item -> user -----------------------------
    fold_kernel<<<64, 256, 0, stream>>>(Wk_i, bk_i, watt_iu, Wc_k, bc_k);
    fold_kernel<<<64, 256, 0, stream>>>(Wv_i, bv_i, wmsg_iu, Wc_v, bc_v);
    gemm_wmma_kernel<0><<<GEMM_BLOCKS, 256, 0, stream>>>(feats_item, Wc_k, bc_k,
        nullptr, nullptr, Kbuf, N_NODE);
    gemm_wmma_kernel<0><<<GEMM_BLOCKS, 256, 0, stream>>>(feats_item, Wc_v, bc_v,
        nullptr, nullptr, Vbuf, N_NODE);
    gemm_wmma_kernel<0><<<GEMM_BLOCKS, 256, 0, stream>>>(feats_user, Wq_u, bq_u,
        nullptr, nullptr, Qbuf, N_NODE);
    init_kernel<<<ELEM_BLOCKS, 256, 0, stream>>>(maxBuf, denomBuf, h_iu, N_NODE);
    score_kernel<<<EDGE_BLOCKS, 256, 0, stream>>>(Qbuf, Kbuf, src_iu, dst_iu, mu_iu,
        scoreBuf, maxBuf);
    agg_kernel<<<EDGE_BLOCKS, 256, 0, stream>>>(Vbuf, src_iu, dst_iu, scoreBuf, maxBuf,
        denomBuf, h_iu);
    norm_h_kernel<<<ELEM_BLOCKS, 256, 0, stream>>>(h_iu, denomBuf, N_NODE);

    // ---------------- node outputs: gated residual GEMM + LayerNorm ---------
    gemm_wmma_kernel<1><<<GEMM_BLOCKS, 256, 0, stream>>>(h_iu, Wa_u, ba_u,
        feats_user, skip_u, out_user, N_NODE);
    gemm_wmma_kernel<1><<<GEMM_BLOCKS, 256, 0, stream>>>(h_ui, Wa_i, ba_i,
        feats_item, skip_i, out_item, N_NODE);
    ln_kernel<<<LN_BLOCKS, 256, 0, stream>>>(out_user, lng_u, lnb_u, N_NODE);
    ln_kernel<<<LN_BLOCKS, 256, 0, stream>>>(out_item, lng_i, lnb_i, N_NODE);
}
